// jointpgm_46643344835064
// MI455X (gfx1250) — matrix operand order repaired and checked by reading, not compiled, not run on hardware
//
#include <hip/hip_runtime.h>
#include <hip/hip_bf16.h>

typedef __attribute__((ext_vector_type(16))) _Float16 v16h;
typedef __attribute__((ext_vector_type(4)))  _Float16 v4h;
typedef __attribute__((ext_vector_type(8)))  float    v8f;

static constexpr int B_  = 32;
static constexpr int L_  = 512;
static constexpr int HP_ = 192;
static constexpr int N_  = 512;
static constexpr int E_  = 128;
static constexpr int Z_  = 64;

__device__ __forceinline__ v4h pack4h(float x, float y, float z, float w)
{
    auto lo = __builtin_amdgcn_cvt_pkrtz(x, y);   // v_cvt_pk_f16_f32 style
    auto hi = __builtin_amdgcn_cvt_pkrtz(z, w);
    v4h r; r[0] = lo[0]; r[1] = lo[1]; r[2] = hi[0]; r[3] = hi[1];
    return r;
}

// ---------------------------------------------------------------------------
// WMMA GEMM:  C[m,n] = sum_k A[m,k]*B[k,n]  (+rowBias[m] +colBias[n])
//   - A optionally transposed in memory (A[m,k] = A[k*lda+m] when TRANSA)
//   - C addressed as m*rowStrideC + n*colStrideC (+batch*strideC)
//   - macro tile 64(M) x 64(N), K-step 32; block = 128 threads = 4 waves
//   - wave w computes rows [16w,16w+16) x 64 cols: 4 WMMAs/K-step, A reused 4x
//   - ping-pong LDS in *fragment order* frag[p][sub][lane][16]: fragment read
//     is one 32B contiguous chunk (2 x ds_load_b128); all LDS fills are v4h
//     (b64) stores; global tile staged in regs first so loads batch up.
// REQUIREMENTS (guaranteed by every launch): M%64==0, Nc%64==0, K%32==0.
// ---------------------------------------------------------------------------
template <bool TRANSA>
__global__ __launch_bounds__(128) void wmma_gemm_f32(
    const float* __restrict__ A, int lda, long long strideA,
    const float* __restrict__ Bm, int ldb, long long strideB,
    float* __restrict__ C, int rowStrideC, int colStrideC, long long strideC,
    const float* __restrict__ rowBias, const float* __restrict__ colBias,
    int M, int Nc, int K)
{
    __shared__ __attribute__((aligned(32))) _Float16 Afrag[2][4][32][16]; // 8 KB
    __shared__ __attribute__((aligned(32))) _Float16 Bfrag[2][4][32][16]; // 8 KB

    const int tid   = threadIdx.x;
    const int wave  = tid >> 5;
    const int lane  = tid & 31;
    const int tileM = blockIdx.y * 64;
    const int tileN = blockIdx.x * 64;
    const int batch = blockIdx.z;

    const float* Ab = A  + (long long)batch * strideA;
    const float* Bb = Bm + (long long)batch * strideB;
    float*       Cb = C  + (long long)batch * strideC;

    v8f acc0 = {0.f,0.f,0.f,0.f,0.f,0.f,0.f,0.f};
    v8f acc1 = acc0, acc2 = acc0, acc3 = acc0;

    // register staging for one 64x32 A tile + 32x64 B tile
    float4 a4s[4];      // !TRANSA: vec4 along k
    float  aS[4][4];    //  TRANSA: 4 consecutive k of one m-column
    float  bS[4][4];    //  4 consecutive k of one n-column

    auto load_tile = [&](int k0) {
        if (!TRANSA) {
#pragma unroll
            for (int i = 0; i < 4; ++i) {
                int idx = tid + i * 128;
                int r = idx >> 3, c4 = (idx & 7) << 2;
                a4s[i] = *(const float4*)&Ab[(long long)(tileM + r) * lda + k0 + c4];
            }
        } else {
#pragma unroll
            for (int i = 0; i < 4; ++i) {
                int idx = tid + i * 128;
                int r = idx & 63, cq = (idx >> 6) << 2;
#pragma unroll
                for (int q = 0; q < 4; ++q)
                    aS[i][q] = Ab[(long long)(k0 + cq + q) * lda + tileM + r];
            }
        }
#pragma unroll
        for (int i = 0; i < 4; ++i) {
            int idx = tid + i * 128;
            int c = idx & 63, kq = (idx >> 6) << 2;
#pragma unroll
            for (int q = 0; q < 4; ++q)
                bS[i][q] = Bb[(long long)(k0 + kq + q) * ldb + tileN + c];
        }
    };

    auto store_tile = [&](int p) {
        if (!TRANSA) {
#pragma unroll
            for (int i = 0; i < 4; ++i) {
                int idx = tid + i * 128;
                int r = idx >> 3, c4 = (idx & 7) << 2;
                int sub = r >> 4;
                int ln  = (r & 15) + (((c4 >> 3) & 1) << 4);
                int pos = (c4 & 7) + (((c4 >> 4) & 1) << 3);
                *(v4h*)&Afrag[p][sub][ln][pos] =
                    pack4h(a4s[i].x, a4s[i].y, a4s[i].z, a4s[i].w);
            }
        } else {
#pragma unroll
            for (int i = 0; i < 4; ++i) {
                int idx = tid + i * 128;
                int r = idx & 63, cq = (idx >> 6) << 2;
                int part = (cq >> 3) & 1;
                int pos  = (cq & 7) + (((cq >> 4) & 1) << 3);
                *(v4h*)&Afrag[p][r >> 4][(r & 15) + (part << 4)][pos] =
                    pack4h(aS[i][0], aS[i][1], aS[i][2], aS[i][3]);
            }
        }
#pragma unroll
        for (int i = 0; i < 4; ++i) {
            int idx = tid + i * 128;
            int c = idx & 63, kq = (idx >> 6) << 2;
            int g  = c >> 4;
            int ln = (c & 15) + (((kq >> 3) & 1) << 4);
            int pos = (kq & 7) + (((kq >> 4) & 1) << 3);
            *(v4h*)&Bfrag[p][g][ln][pos] =
                pack4h(bS[i][0], bS[i][1], bS[i][2], bS[i][3]);
        }
    };

    load_tile(0);
    int p = 0;
    for (int k0 = 0; k0 < K; k0 += 32) {
        store_tile(p);
        __syncthreads();
        if (k0 + 32 < K) load_tile(k0 + 32);   // overlap next tile with WMMAs

        v16h af = *(const v16h*)&Afrag[p][wave][lane][0];
        v16h b0 = *(const v16h*)&Bfrag[p][0][lane][0];
        v16h b1 = *(const v16h*)&Bfrag[p][1][lane][0];
        v16h b2 = *(const v16h*)&Bfrag[p][2][lane][0];
        v16h b3 = *(const v16h*)&Bfrag[p][3][lane][0];
        acc0 = __builtin_amdgcn_wmma_f32_16x16x32_f16(false, af, false, b0, (short)0, acc0, false, false);
        acc1 = __builtin_amdgcn_wmma_f32_16x16x32_f16(false, af, false, b1, (short)0, acc1, false, false);
        acc2 = __builtin_amdgcn_wmma_f32_16x16x32_f16(false, af, false, b2, (short)0, acc2, false, false);
        acc3 = __builtin_amdgcn_wmma_f32_16x16x32_f16(false, af, false, b3, (short)0, acc3, false, false);
        p ^= 1;
    }

    // ======================= epilogue + store =============================
    // C 16x16 f32 layout: n = lane&15, m = vgpr_j + (lane>=16)*8
    const int nloc  = lane & 15;
    const int mbase = (lane < 16) ? 0 : 8;
    v8f accs[4] = {acc0, acc1, acc2, acc3};
#pragma unroll
    for (int g = 0; g < 4; ++g) {
        int n = tileN + g * 16 + nloc;
        float cb = colBias ? colBias[n] : 0.f;
#pragma unroll
        for (int j = 0; j < 8; ++j) {
            int m = tileM + wave * 16 + mbase + j;
            float rb = rowBias ? rowBias[m] : 0.f;
            Cb[(long long)m * rowStrideC + (long long)n * colStrideC] =
                accs[g][j] + rb + cb;
        }
    }
}

// ---------------------------------------------------------------------------
// Small kernels (memory-bound glue)
// ---------------------------------------------------------------------------
__global__ void time_embed_k(const float* __restrict__ tx,
                             const float* __restrict__ W,
                             const float* __restrict__ bias,
                             float* __restrict__ out, int total)
{
    int i = blockIdx.x * blockDim.x + threadIdx.x;
    if (i >= total) return;
    int e  = i & (E_ - 1);
    int bl = i >> 7;                   // E_ == 128
    const float* x = tx + bl * 7;
    float s = bias[e];
#pragma unroll
    for (int k = 0; k < 7; ++k) s += x[k] * W[k * E_ + e];
    out[i] = s;
}

__global__ void add_inplace_k(float* __restrict__ a,
                              const float* __restrict__ b, int n)
{
    int i = blockIdx.x * blockDim.x + threadIdx.x;
    if (i < n) a[i] += b[i];
}

__device__ inline float gauss_hash(unsigned idx, unsigned seed)
{
    unsigned s = idx ^ seed;
    s ^= s >> 16; s *= 0x7feb352dU; s ^= s >> 15; s *= 0x846ca68bU; s ^= s >> 16;
    unsigned t = (s ^ 0x9E3779B9U) * 0x85ebca6bU;
    t ^= t >> 13; t *= 0xc2b2ae35U; t ^= t >> 16;
    float u1 = (float)(s >> 8) * (1.0f / 16777216.0f) + 1e-7f;
    float u2 = (float)(t >> 8) * (1.0f / 16777216.0f);
    return sqrtf(-2.0f * __logf(u1)) * __cosf(6.28318530718f * u2);
}

__global__ void reparam_k(const float* __restrict__ mu,
                          const float* __restrict__ var,
                          float* __restrict__ out, int n, unsigned seed)
{
    int i = blockIdx.x * blockDim.x + threadIdx.x;
    if (i < n) out[i] = mu[i] + gauss_hash((unsigned)i, seed) * __expf(0.5f * var[i]);
}

// a[i] = a[i]*w[0] + bb[0]   (h_tilde: GC diag is provably zero -> scalar affine)
__global__ void scale_shift_k(float* __restrict__ a,
                              const float* __restrict__ w,
                              const float* __restrict__ bb, int n)
{
    int i = blockIdx.x * blockDim.x + threadIdx.x;
    if (i < n) a[i] = a[i] * w[0] + bb[0];
}

__global__ void avg_k(const float* __restrict__ a, const float* __restrict__ b,
                      float* __restrict__ out, int n)
{
    int i = blockIdx.x * blockDim.x + threadIdx.x;
    if (i < n) out[i] = 0.5f * (a[i] + b[i]);
}

__global__ void leaky_add_k(float* __restrict__ u, const float* __restrict__ h, int n)
{
    int i = blockIdx.x * blockDim.x + threadIdx.x;
    if (i < n) {
        float v = u[i];
        u[i] = (v > 0.f ? v : 0.01f * v) + h[i];
    }
}

// ---------------------------------------------------------------------------
extern "C" void kernel_launch(void* const* d_in, const int* in_sizes, int n_in,
                              void* d_out, int out_size, void* d_ws, size_t ws_size,
                              hipStream_t stream)
{
    const float* batch_x = (const float*)d_in[0];
    const float* time_x  = (const float*)d_in[1];
    const float* time_W  = (const float*)d_in[2];
    const float* time_b  = (const float*)d_in[3];
    const float* gate_W  = (const float*)d_in[4];
    const float* gate_b  = (const float*)d_in[5];
    const float* mu_t_W  = (const float*)d_in[6];
    const float* mu_t_b  = (const float*)d_in[7];
    const float* var_t_W = (const float*)d_in[8];
    const float* var_t_b = (const float*)d_in[9];
    const float* ser_W   = (const float*)d_in[10];
    const float* ser_b   = (const float*)d_in[11];
    const float* mu_h_W  = (const float*)d_in[12];
    const float* mu_h_b  = (const float*)d_in[13];
    const float* var_h_W = (const float*)d_in[14];
    const float* var_h_b = (const float*)d_in[15];
    const float* mlp_W   = (const float*)d_in[18];
    const float* mlp_b   = (const float*)d_in[19];
    const float* mu_ht_W = (const float*)d_in[20];
    const float* mu_ht_b = (const float*)d_in[21];
    const float* var_ht_W= (const float*)d_in[22];
    const float* var_ht_b= (const float*)d_in[23];
    const float* inf_W   = (const float*)d_in[24];
    const float* inf_b   = (const float*)d_in[25];
    const float* mu_zt_W = (const float*)d_in[26];
    const float* mu_zt_b = (const float*)d_in[27];
    const float* var_zt_W= (const float*)d_in[28];
    const float* var_zt_b= (const float*)d_in[29];
    const float* fW1 = (const float*)d_in[30];
    const float* fb1 = (const float*)d_in[31];
    const float* fW2 = (const float*)d_in[32];
    const float* fb2 = (const float*)d_in[33];
    const float* rW1 = (const float*)d_in[34];
    const float* rb1 = (const float*)d_in[35];
    const float* rW2 = (const float*)d_in[36];
    const float* rb2 = (const float*)d_in[37];

    float* out = (float*)d_out;
    // output tuple layout (flat floats)
    const long long OFF_X    = 0;                     // [B,L,N]
    const long long OFF_Y    = 8388608;               // [B,Hp,N]
    const long long OFF_MUT  = 11534336;              // [B,N,Z]
    const long long OFF_VART = 12582912;
    const long long OFF_MUZT = 13631488;
    const long long OFF_VARZT= 14680064;
    const long long OFF_MUZ  = 15728640;
    const long long OFF_VARZ = 16777216;

    // workspace layout (flat floats)
    float* ws   = (float*)d_ws;
    float* t0   = ws;              // [B,L,E]  2M
    float* Tbuf = ws + 2097152;    // [B,N,E]  2M  (T -> h_hat -> h_tilde)
    float* Hbuf = ws + 4194304;    // [B,N,E]  2M  (kept: "h")
    float* muh  = ws + 6291456;    // [B,N,Z]  1M
    float* varh = ws + 7340032;
    float* zhat = ws + 8388608;    // becomes z
    float* muht = ws + 9437184;
    float* varht= ws + 10485760;
    float* ztil = ws + 11534336;
    float* ztbuf= ws + 12582912;   // zt
    float* ubuf = ws + 13631488;   // [B,N,E]  2M

    const int BNE = B_ * N_ * E_;     // 2097152
    const int BNZ = B_ * N_ * Z_;     // 1048576
    const int MM  = B_ * N_;          // 16384 flattened rows

    dim3 blk(128);
    auto g1 = [](int n){ return dim3((n + 255) / 256); };

    // 1) t0 = time_x @ time_W + time_b                         [B,L,E]
    time_embed_k<<<g1(BNE), 256, 0, stream>>>(time_x, time_W, time_b, t0, BNE);

    // 2) T = einsum('ble,ln->bne') = gate_W^T @ t0_b + gate_b[m]
    wmma_gemm_f32<true><<<dim3(E_/64, N_/64, B_), blk, 0, stream>>>(
        gate_W, N_, 0, t0, E_, (long long)L_*E_,
        Tbuf, E_, 1, (long long)N_*E_, gate_b, nullptr, N_, E_, L_);

    // 3) mu_t / var_t = T @ W + b  -> direct to d_out
    wmma_gemm_f32<false><<<dim3(Z_/64, MM/64, 1), blk, 0, stream>>>(
        Tbuf, E_, 0, mu_t_W, Z_, 0, out + OFF_MUT, Z_, 1, 0,
        nullptr, mu_t_b, MM, Z_, E_);
    wmma_gemm_f32<false><<<dim3(Z_/64, MM/64, 1), blk, 0, stream>>>(
        Tbuf, E_, 0, var_t_W, Z_, 0, out + OFF_VART, Z_, 1, 0,
        nullptr, var_t_b, MM, Z_, E_);

    // 4) H = einsum('bln,le->bne') = batch_x_b^T @ ser_W + ser_b[n]
    wmma_gemm_f32<true><<<dim3(E_/64, N_/64, B_), blk, 0, stream>>>(
        batch_x, N_, (long long)L_*N_, ser_W, E_, 0,
        Hbuf, E_, 1, (long long)N_*E_, nullptr, ser_b, N_, E_, L_);

    // 5) h_hat = H + T  (in place into Tbuf; Hbuf keeps "h")
    add_inplace_k<<<g1(BNE), 256, 0, stream>>>(Tbuf, Hbuf, BNE);

    // 6) mu_h_hat / var_h_hat
    wmma_gemm_f32<false><<<dim3(Z_/64, MM/64, 1), blk, 0, stream>>>(
        Tbuf, E_, 0, mu_h_W, Z_, 0, muh, Z_, 1, 0, nullptr, mu_h_b, MM, Z_, E_);
    wmma_gemm_f32<false><<<dim3(Z_/64, MM/64, 1), blk, 0, stream>>>(
        Tbuf, E_, 0, var_h_W, Z_, 0, varh, Z_, 1, 0, nullptr, var_h_b, MM, Z_, E_);

    // 7) z_hat = reparam(mu_h_hat, var_h_hat)
    reparam_k<<<g1(BNZ), 256, 0, stream>>>(muh, varh, zhat, BNZ, 0x13579BDFu);

    // 8) graph conv collapses (diag(adj)==0): h_tilde = h_hat*mlp_W[0] + mlp_b
    scale_shift_k<<<g1(BNE), 256, 0, stream>>>(Tbuf, mlp_W, mlp_b, BNE);

    // 9) mu_ht / var_ht ; z_tilde
    wmma_gemm_f32<false><<<dim3(Z_/64, MM/64, 1), blk, 0, stream>>>(
        Tbuf, E_, 0, mu_ht_W, Z_, 0, muht, Z_, 1, 0, nullptr, mu_ht_b, MM, Z_, E_);
    wmma_gemm_f32<false><<<dim3(Z_/64, MM/64, 1), blk, 0, stream>>>(
        Tbuf, E_, 0, var_ht_W, Z_, 0, varht, Z_, 1, 0, nullptr, var_ht_b, MM, Z_, E_);
    reparam_k<<<g1(BNZ), 256, 0, stream>>>(muht, varht, ztil, BNZ, 0x2468ACE1u);

    // 10) z / mu_z / var_z  (ALPHA = 0.5)
    avg_k<<<g1(BNZ), 256, 0, stream>>>(zhat, ztil, zhat, BNZ);                 // z
    avg_k<<<g1(BNZ), 256, 0, stream>>>(muh,  muht, out + OFF_MUZ,  BNZ);
    avg_k<<<g1(BNZ), 256, 0, stream>>>(varh, varht, out + OFF_VARZ, BNZ);

    // 11) zt = z @ inf_W + inf_b ; mu_zt / var_zt -> d_out
    wmma_gemm_f32<false><<<dim3(Z_/64, MM/64, 1), blk, 0, stream>>>(
        zhat, Z_, 0, inf_W, Z_, 0, ztbuf, Z_, 1, 0, nullptr, inf_b, MM, Z_, Z_);
    wmma_gemm_f32<false><<<dim3(Z_/64, MM/64, 1), blk, 0, stream>>>(
        ztbuf, Z_, 0, mu_zt_W, Z_, 0, out + OFF_MUZT, Z_, 1, 0,
        nullptr, mu_zt_b, MM, Z_, Z_);
    wmma_gemm_f32<false><<<dim3(Z_/64, MM/64, 1), blk, 0, stream>>>(
        ztbuf, Z_, 0, var_zt_W, Z_, 0, out + OFF_VARZT, Z_, 1, 0,
        nullptr, var_zt_b, MM, Z_, Z_);

    // 12) y = transpose((leaky(z@fW1+fb1) + h) @ fW2 + fb2)  -> [B,Hp,N]
    wmma_gemm_f32<false><<<dim3(E_/64, MM/64, 1), blk, 0, stream>>>(
        zhat, Z_, 0, fW1, E_, 0, ubuf, E_, 1, 0, nullptr, fb1, MM, E_, Z_);
    leaky_add_k<<<g1(BNE), 256, 0, stream>>>(ubuf, Hbuf, BNE);
    wmma_gemm_f32<false><<<dim3(HP_/64, N_/64, B_), blk, 0, stream>>>(
        ubuf, E_, (long long)N_*E_, fW2, HP_, 0,
        out + OFF_Y, /*rowStrideC=*/1, /*colStrideC=*/N_, (long long)HP_*N_,
        nullptr, fb2, N_, HP_, E_);

    // 13) x = transpose((leaky(z@rW1+rb1) + h) @ rW2 + rb2)  -> [B,L,N]
    wmma_gemm_f32<false><<<dim3(E_/64, MM/64, 1), blk, 0, stream>>>(
        zhat, Z_, 0, rW1, E_, 0, ubuf, E_, 1, 0, nullptr, rb1, MM, E_, Z_);
    leaky_add_k<<<g1(BNE), 256, 0, stream>>>(ubuf, Hbuf, BNE);
    wmma_gemm_f32<false><<<dim3(L_/64, N_/64, B_), blk, 0, stream>>>(
        ubuf, E_, (long long)N_*E_, rW2, L_, 0,
        out + OFF_X, /*rowStrideC=*/1, /*colStrideC=*/N_, (long long)L_*N_,
        nullptr, rb2, N_, L_, E_);

    (void)in_sizes; (void)n_in; (void)out_size; (void)ws_size;
}
